// TransformerModel_1726576853024
// MI455X (gfx1250) — matrix operand order repaired and checked by reading, not compiled, not run on hardware
//
#include <hip/hip_runtime.h>

// ---------------- types ----------------
typedef __bf16 bf16;
typedef float  v8f    __attribute__((ext_vector_type(8)));
typedef bf16   v16bf  __attribute__((ext_vector_type(16)));
typedef bf16   bf16x8 __attribute__((ext_vector_type(8)));

// ---------------- model dims ----------------
#define B_   2
#define T_   1024
#define NT_  30
#define H_   1024
#define L_   8
#define F_   2816
#define NH_  16
#define R_   16
#define RP_  32                 // rank padded to one K-chunk
#define HD_  64
#define M_   (B_ * T_)          // 2048 token rows
#define NC_  (NT_ + 1)          // 31 head outputs
#define LORA_SCALE_ 2.0f
#define CLAMP_ 10.0f
#define EPS_ 1e-6f

enum { GF_ACC = 1, GF_OUTB = 2, GF_WRITEF = 4, GF_BIAS = 8 };

// async copy of 16B to LDS (gfx1250 VGLOBAL async op, tracked by ASYNCcnt)
#define ASYNC_B128(ldsoff, gptr)                                              \
    asm volatile("global_load_async_to_lds_b128 %0, %1, off"                  \
                 :: "v"(ldsoff),                                              \
                    "v"((unsigned long long)(size_t)(gptr))                   \
                 : "memory")
#define WAIT_ASYNC() asm volatile("s_wait_asynccnt 0x0" ::: "memory")
#define WAIT_DS()    asm volatile("s_wait_dscnt 0x0" ::: "memory")

// =====================================================================
// Generic WMMA bf16 GEMM:
//   C[M x N](+=) alpha * (A[M x K]bf16 @ B[Kb x N]fp32) (+bias)
// A row stride = lda (K always a multiple of 32 -> pure async staging).
// B element (k,n) read at Bw[k*ldbK + n*ldbN]; rows k >= Kb are zero
// (pairs with zero-padded A columns for the rank-16 LoRA matmuls).
// C row stride = ldc. Block = 256 thr (8 waves), tile 128x64, K-chunk 32,
// double-buffered LDS; A staged via global_load_async_to_lds_b128.
// =====================================================================
__global__ __launch_bounds__(256)
void k_gemm(const bf16* __restrict__ A, const float* __restrict__ Bw,
            const float* __restrict__ bias, float* __restrict__ Cf,
            bf16* __restrict__ Cb, int N, int K, int Kb, int lda, int ldc,
            long ldbK, long ldbN, float alpha, int flags)
{
    __shared__ bf16 As[2][128][48];   // row stride 96B (16B aligned, bank-spread)
    __shared__ bf16 Bs[2][64][48];    // stored transposed: Bs[n][k]

    const int tid  = threadIdx.x;
    const int lane = tid & 31;
    const int wave = tid >> 5;          // 0..7 -> 16-row strip of the 128-row tile
    const int mB   = blockIdx.y * 128;
    const int nB   = blockIdx.x * 64;

    const int arow = tid >> 1;          // 0..127
    const int ac   = (tid & 1) * 16;    // 0 / 16
    const int brow = tid >> 3;          // 0..31  (k within chunk)
    const int bn   = (tid & 7) * 8;     // n start

    v8f acc[4];
    for (int t = 0; t < 4; ++t)
        for (int i = 0; i < 8; ++i) acc[t][i] = 0.f;

    const int nTiles = K >> 5;          // K is always a multiple of 32

    auto stage = [&](int kt, int buf) {
        const int k0 = kt * 32;
        // ---- A tile 128x32 bf16: async copy straight into LDS ----
        const bf16* src = A + (size_t)(mB + arow) * lda + k0 + ac;
        const unsigned d0 = (unsigned)(size_t)&As[buf][arow][ac];
        const unsigned d1 = (unsigned)(size_t)&As[buf][arow][ac + 8];
        ASYNC_B128(d0, src);
        ASYNC_B128(d1, src + 8);
        // ---- B tile 32x64 fp32 -> bf16, transposed, branchless guards ----
        const int  kk = k0 + brow;
        const int  kc = kk < Kb ? kk : 0;
        for (int i = 0; i < 8; ++i) {
            const int  n  = bn + i;
            const int  nc = (nB + n) < N ? (nB + n) : 0;
            const float v = Bw[(size_t)kc * ldbK + (size_t)nc * ldbN];
            const bool ok = (kk < Kb) && ((nB + n) < N);
            Bs[buf][n][brow] = (bf16)(ok ? v : 0.f);
        }
    };

    stage(0, 0);

    for (int kt = 0; kt < nTiles; ++kt) {
        const int buf = kt & 1;
        WAIT_ASYNC();                    // our async tile writes are in LDS
        __syncthreads();                 // everyone's tile kt is visible
        if (kt + 1 < nTiles) stage(kt + 1, buf ^ 1);
        if (kt + 2 < nTiles)             // warm L2 for the tile after next
            __builtin_prefetch(Bw + (size_t)(kt * 32 + 64 + brow) * ldbK +
                                   (size_t)(nB + bn) * ldbN, 0, 1);

        // ---- A fragment (16x32, ISA 16-bit A layout) ----
        v16bf af;
        {
            const int r   = wave * 16 + (lane & 15);
            const int klo = (lane < 16) ? 0 : 8;
            bf16x8 lo = *(const bf16x8*)&As[buf][r][klo];
            bf16x8 hi = *(const bf16x8*)&As[buf][r][klo + 16];
            for (int i = 0; i < 8; ++i) { af[i] = lo[i]; af[i + 8] = hi[i]; }
        }
        // ---- 4 B fragments (32x16) + 4 WMMAs ----
        const int klo = (lane < 16) ? 0 : 8;
        for (int nt = 0; nt < 4; ++nt) {
            const int n = nt * 16 + (lane & 15);
            v16bf bfr;
            bf16x8 b0 = *(const bf16x8*)&Bs[buf][n][klo];
            bf16x8 b1 = *(const bf16x8*)&Bs[buf][n][klo + 16];
            for (int i = 0; i < 8; ++i) { bfr[i] = b0[i]; bfr[i + 8] = b1[i]; }
            acc[nt] = __builtin_amdgcn_wmma_f32_16x16x32_bf16(false, af, false, bfr,
                                                              (short)0, acc[nt],
                                                              false, false);
        }
        __syncthreads();                 // tile kt fully consumed before reuse
    }

    // ---- epilogue: C layout = VGPR g, lanes<16 -> M=g, lanes>=16 -> M=g+8 ----
    const int mrow0 = mB + wave * 16 + ((lane >= 16) ? 8 : 0);
    for (int nt = 0; nt < 4; ++nt) {
        const int col = nB + nt * 16 + (lane & 15);
        if (col < N) {
            for (int g = 0; g < 8; ++g) {
                const size_t idx = (size_t)(mrow0 + g) * ldc + col;
                float v = alpha * acc[nt][g];
                if (flags & GF_ACC)    v += Cf[idx];
                if (flags & GF_BIAS)   v += bias[col];
                if (flags & GF_WRITEF) Cf[idx] = v;
                if (flags & GF_OUTB)   Cb[idx] = (bf16)v;
            }
        }
    }
}

// =====================================================================
// Flash attention, causal. Block = (b,h, 128 queries); wave = 16 queries.
// scores via 2 chained WMMAs over HD=64; P.V via 4 WMMAs per 32-key step.
// K tile staged asynchronously; V transposed through VGPRs.
// =====================================================================
__global__ __launch_bounds__(256)
void k_attn(const bf16* __restrict__ Q, const bf16* __restrict__ Kg,
            const bf16* __restrict__ Vg, bf16* __restrict__ O)
{
    __shared__ bf16 Ks[32][80];       // [key][d]   row stride 160B
    __shared__ bf16 Vt[64][48];       // [d][key]   row stride  96B
    __shared__ bf16 Pl[8][16][48];    // per-wave P tile [q][key32]

    const int tid = threadIdx.x, lane = tid & 31, w = tid >> 5;
    const int qb = blockIdx.x, h = blockIdx.y, b = blockIdx.z;
    const int q0 = qb * 128 + w * 16;
    const int half8 = (lane >= 16) ? 8 : 0;

    // ---- Q fragments (d-chunks 0-31 / 32-63) ----
    v16bf qf0, qf1;
    {
        const int r   = q0 + (lane & 15);
        const int klo = (lane < 16) ? 0 : 8;
        const bf16* qp = Q + (size_t)(b * T_ + r) * H_ + h * HD_;
        bf16x8 a0 = *(const bf16x8*)(qp + klo);
        bf16x8 a1 = *(const bf16x8*)(qp + klo + 16);
        bf16x8 a2 = *(const bf16x8*)(qp + 32 + klo);
        bf16x8 a3 = *(const bf16x8*)(qp + 32 + klo + 16);
        for (int i = 0; i < 8; ++i) {
            qf0[i] = a0[i]; qf0[i + 8] = a1[i];
            qf1[i] = a2[i]; qf1[i + 8] = a3[i];
        }
    }

    float mrun[8], lrun[8];
    v8f acc[4];
    for (int g = 0; g < 8; ++g) { mrun[g] = -1e30f; lrun[g] = 0.f; }
    for (int dt = 0; dt < 4; ++dt)
        for (int g = 0; g < 8; ++g) acc[dt][g] = 0.f;

    const int nKB = (qb + 1) * 4;          // 32 keys per step, causal bound
    for (int kb = 0; kb < nKB; ++kb) {
        const int kbase = kb * 32;
        // ---- stage K (async, row major) and V (transposed) ----
        {
            const int kk = tid >> 3, d0 = (tid & 7) * 8;
            const size_t src = (size_t)(b * T_ + kbase + kk) * H_ + h * HD_ + d0;
            const unsigned kd = (unsigned)(size_t)&Ks[kk][d0];
            ASYNC_B128(kd, Kg + src);
            bf16x8 vv = *(const bf16x8*)(Vg + src);
            for (int i = 0; i < 8; ++i) Vt[d0 + i][kk] = vv[i];
        }
        WAIT_ASYNC();
        __syncthreads();

        if (kbase <= q0 + 15) {            // wave-uniform: EXEC stays full
            // ---- S = Q.K^T for two 16-key subtiles ----
            v8f sc[2];
            for (int nk = 0; nk < 2; ++nk) {
                const int key = nk * 16 + (lane & 15);
                const int klo = (lane < 16) ? 0 : 8;
                v16bf kf0, kf1;
                bf16x8 b0 = *(const bf16x8*)&Ks[key][klo];
                bf16x8 b1 = *(const bf16x8*)&Ks[key][klo + 16];
                bf16x8 b2 = *(const bf16x8*)&Ks[key][32 + klo];
                bf16x8 b3 = *(const bf16x8*)&Ks[key][32 + klo + 16];
                for (int i = 0; i < 8; ++i) {
                    kf0[i] = b0[i]; kf0[i + 8] = b1[i];
                    kf1[i] = b2[i]; kf1[i + 8] = b3[i];
                }
                v8f s;
                for (int i = 0; i < 8; ++i) s[i] = 0.f;
                s = __builtin_amdgcn_wmma_f32_16x16x32_bf16(false, qf0, false, kf0,
                                                            (short)0, s, false, false);
                s = __builtin_amdgcn_wmma_f32_16x16x32_bf16(false, qf1, false, kf1,
                                                            (short)0, s, false, false);
                sc[nk] = s;
            }
            // ---- scale + causal mask ----
            for (int nk = 0; nk < 2; ++nk) {
                const int kidx = kbase + nk * 16 + (lane & 15);
                for (int g = 0; g < 8; ++g) {
                    const int qidx = q0 + g + half8;
                    sc[nk][g] = (kidx <= qidx) ? sc[nk][g] * 0.125f : -1e9f;
                }
            }
            // ---- online softmax (row lives on 16 lanes of one half) ----
            float fg[8];
            for (int g = 0; g < 8; ++g) {
                float rm = fmaxf(sc[0][g], sc[1][g]);
                for (int m = 1; m < 16; m <<= 1) rm = fmaxf(rm, __shfl_xor(rm, m, 32));
                const float mn = fmaxf(mrun[g], rm);
                const float f  = __expf(mrun[g] - mn);
                const float p0 = __expf(sc[0][g] - mn);
                const float p1 = __expf(sc[1][g] - mn);
                sc[0][g] = p0; sc[1][g] = p1;
                float rs = p0 + p1;
                for (int m = 1; m < 16; m <<= 1) rs += __shfl_xor(rs, m, 32);
                lrun[g] = lrun[g] * f + rs;
                mrun[g] = mn;
                fg[g]   = f;
            }
            for (int dt = 0; dt < 4; ++dt)
                for (int g = 0; g < 8; ++g) acc[dt][g] *= fg[g];

            // ---- C-layout -> A-layout via wave-private LDS ----
            for (int g = 0; g < 8; ++g) {
                const int m = g + half8;
                Pl[w][m][lane & 15]        = (bf16)sc[0][g];
                Pl[w][m][16 + (lane & 15)] = (bf16)sc[1][g];
            }
            WAIT_DS();
            v16bf pf;
            {
                const int r   = lane & 15;
                const int klo = (lane < 16) ? 0 : 8;
                bf16x8 lo = *(const bf16x8*)&Pl[w][r][klo];
                bf16x8 hi = *(const bf16x8*)&Pl[w][r][klo + 16];
                for (int i = 0; i < 8; ++i) { pf[i] = lo[i]; pf[i + 8] = hi[i]; }
            }
            // ---- acc += P.V ----
            for (int dt = 0; dt < 4; ++dt) {
                const int d   = dt * 16 + (lane & 15);
                const int klo = (lane < 16) ? 0 : 8;
                v16bf vf;
                bf16x8 v0 = *(const bf16x8*)&Vt[d][klo];
                bf16x8 v1 = *(const bf16x8*)&Vt[d][klo + 16];
                for (int i = 0; i < 8; ++i) { vf[i] = v0[i]; vf[i + 8] = v1[i]; }
                acc[dt] = __builtin_amdgcn_wmma_f32_16x16x32_bf16(false, pf, false, vf,
                                                                  (short)0, acc[dt],
                                                                  false, false);
            }
        }
        __syncthreads();
    }

    // ---- normalize + write (B,T,H) bf16 ----
    for (int dt = 0; dt < 4; ++dt) {
        const int d = h * HD_ + dt * 16 + (lane & 15);
        for (int g = 0; g < 8; ++g) {
            const int trow = q0 + g + half8;
            O[(size_t)(b * T_ + trow) * H_ + d] = (bf16)(acc[dt][g] / lrun[g]);
        }
    }
}

// =====================================================================
// Elementwise kernels
// =====================================================================
__global__ __launch_bounds__(256)
void k_zero_bf16(bf16* __restrict__ p, int n)
{
    const int i = blockIdx.x * 256 + threadIdx.x;
    if (i < n) p[i] = (bf16)0.0f;
}

__global__ __launch_bounds__(256)
void k_embed(const float* __restrict__ x, const float* __restrict__ W,
             const float* __restrict__ bias, float* __restrict__ h)
{
    __shared__ float xs[NT_];
    const int row = blockIdx.x;
    if (threadIdx.x < NT_) xs[threadIdx.x] = x[(size_t)row * NT_ + threadIdx.x];
    __syncthreads();
    for (int c = threadIdx.x; c < H_; c += 256) {
        float a = bias[c];
        for (int i = 0; i < NT_; ++i) a += xs[i] * W[(size_t)i * H_ + c];
        h[(size_t)row * H_ + c] = a;
    }
}

__global__ __launch_bounds__(256)
void k_rms(const float* __restrict__ x, const float* __restrict__ w,
           bf16* __restrict__ o)
{
    __shared__ float red[256];
    const int row = blockIdx.x;
    const float* xr = x + (size_t)row * H_;
    float s = 0.f;
    for (int i = threadIdx.x; i < H_; i += 256) { float v = xr[i]; s += v * v; }
    red[threadIdx.x] = s;
    __syncthreads();
    for (int st = 128; st > 0; st >>= 1) {
        if (threadIdx.x < st) red[threadIdx.x] += red[threadIdx.x + st];
        __syncthreads();
    }
    const float rs = rsqrtf(red[0] * (1.0f / H_) + EPS_);
    for (int i = threadIdx.x; i < H_; i += 256)
        o[(size_t)row * H_ + i] = (bf16)(xr[i] * rs * w[i]);
}

__global__ __launch_bounds__(256)
void k_rope(const float* __restrict__ q, const float* __restrict__ k,
            const float* __restrict__ v, bf16* __restrict__ qo,
            bf16* __restrict__ ko, bf16* __restrict__ vo)
{
    const int row = blockIdx.x;               // b*T + t
    const int t   = row & (T_ - 1);
    for (int idx = threadIdx.x; idx < H_; idx += 256) {
        const int d  = idx & (HD_ - 1);
        const int hb = idx & ~(HD_ - 1);
        const int dp = (d < 32) ? d : d - 32;
        const float inv = __powf(10000.f, -(float)dp * (1.f / 32.f));
        float sn, cs;
        __sincosf((float)t * inv, &sn, &cs);
        const size_t base = (size_t)row * H_;
        float qv, kv;
        if (d < 32) {
            qv = q[base + idx] * cs - q[base + hb + d + 32] * sn;
            kv = k[base + idx] * cs - k[base + hb + d + 32] * sn;
        } else {
            qv = q[base + idx] * cs + q[base + hb + dp] * sn;
            kv = k[base + idx] * cs + k[base + hb + dp] * sn;
        }
        qo[base + idx] = (bf16)qv;
        ko[base + idx] = (bf16)kv;
        vo[base + idx] = (bf16)v[base + idx];
    }
}

__global__ __launch_bounds__(256)
void k_silumul(const float* __restrict__ g, const float* __restrict__ u,
               bf16* __restrict__ o, int n)
{
    const int i = blockIdx.x * 256 + threadIdx.x;
    if (i < n) {
        const float gv = g[i];
        o[i] = (bf16)((gv / (1.f + __expf(-gv))) * u[i]);
    }
}

__global__ __launch_bounds__(32)
void k_head(const float* __restrict__ logits, const float* __restrict__ temp,
            float* __restrict__ out)
{
    const int row = blockIdx.x, c = threadIdx.x;
    float lc = -1e30f;
    if (c < NC_) {
        const float v = logits[(size_t)row * NC_ + c];
        lc = fminf(fmaxf(v, -CLAMP_), CLAMP_);
    }
    const float z = lc * (1.f / temp[0]);
    float mx = z;
    for (int m = 16; m; m >>= 1) mx = fmaxf(mx, __shfl_xor(mx, m, 32));
    const float e = (c < NC_) ? __expf(z - mx) : 0.f;
    float sm = e;
    for (int m = 16; m; m >>= 1) sm += __shfl_xor(sm, m, 32);
    if (c < NC_) {
        out[(size_t)row * NC_ + c] = e / sm;                      // weights
        out[(size_t)M_ * NC_ + (size_t)row * NC_ + c] = lc;       // logits
    }
}

// =====================================================================
// Host orchestration
// =====================================================================
static inline void gemm(hipStream_t s, const bf16* A, const float* Bw,
                        const float* bias, float* Cf, bf16* Cb,
                        int N, int K, int Kb, int lda, int ldc,
                        long ldbK, long ldbN, float alpha, int flags)
{
    dim3 grid((N + 63) / 64, M_ / 128);
    k_gemm<<<grid, 256, 0, s>>>(A, Bw, bias, Cf, Cb, N, K, Kb, lda, ldc,
                                ldbK, ldbN, alpha, flags);
}

extern "C" void kernel_launch(void* const* d_in, const int* in_sizes, int n_in,
                              void* d_out, int out_size, void* d_ws, size_t ws_size,
                              hipStream_t stream)
{
    (void)in_sizes; (void)n_in; (void)out_size; (void)ws_size;
    const float* x      = (const float*)d_in[0];
    const float* temp   = (const float*)d_in[1];
    const float* stateW = (const float*)d_in[2];
    const float* stateb = (const float*)d_in[3];
    const float* ln1    = (const float*)d_in[4];
    const float* ln2    = (const float*)d_in[5];
    const float* Wqkv   = (const float*)d_in[6];
    const float* Aqkv   = (const float*)d_in[7];
    const float* Bqkv   = (const float*)d_in[8];
    const float* Wo     = (const float*)d_in[9];
    const float* Ao     = (const float*)d_in[10];
    const float* Bo     = (const float*)d_in[11];
    const float* Wgu    = (const float*)d_in[12];
    const float* Agu    = (const float*)d_in[13];
    const float* Bgu    = (const float*)d_in[14];
    const float* Wd     = (const float*)d_in[15];
    const float* Ad     = (const float*)d_in[16];
    const float* Bd     = (const float*)d_in[17];
    const float* lnf    = (const float*)d_in[18];
    const float* headW  = (const float*)d_in[19];
    const float* headb  = (const float*)d_in[20];

    char* p = (char*)d_ws;
    size_t off = 0;
    auto take = [&](size_t bytes) -> void* {
        void* r = p + off;
        off += (bytes + 255) & ~(size_t)255;
        return r;
    };
    float* h      = (float*)take((size_t)M_ * H_ * 4);
    bf16*  hnb    = (bf16*) take((size_t)M_ * H_ * 2);
    float* qk0    = (float*)take((size_t)M_ * H_ * 4);
    float* qk1    = (float*)take((size_t)M_ * H_ * 4);
    float* qk2    = (float*)take((size_t)M_ * H_ * 4);
    bf16*  qb     = (bf16*) take((size_t)M_ * H_ * 2);
    bf16*  kbuf   = (bf16*) take((size_t)M_ * H_ * 2);
    bf16*  vbuf   = (bf16*) take((size_t)M_ * H_ * 2);
    bf16*  attb   = (bf16*) take((size_t)M_ * H_ * 2);
    bf16*  tmpb   = (bf16*) take((size_t)M_ * RP_ * 2);   // rank-padded LoRA tmp
    float* gu0    = (float*)take((size_t)M_ * F_ * 4);
    float* gu1    = (float*)take((size_t)M_ * F_ * 4);
    bf16*  actb   = (bf16*) take((size_t)M_ * F_ * 2);
    float* logits = (float*)take((size_t)M_ * NC_ * 4);

    // zero the padded LoRA tmp once per call (keeps pad columns at 0)
    k_zero_bf16<<<(M_ * RP_ + 255) / 256, 256, 0, stream>>>(tmpb, M_ * RP_);
    k_embed<<<M_, 256, 0, stream>>>(x, stateW, stateb, h);

    float* qks[3] = {qk0, qk1, qk2};
    float* gus[2] = {gu0, gu1};

    for (int l = 0; l < L_; ++l) {
        k_rms<<<M_, 256, 0, stream>>>(h, ln1 + (size_t)l * H_, hnb);
        for (int kk = 0; kk < 3; ++kk) {
            const float* W  = Wqkv + (size_t)(l * 3 + kk) * H_ * H_;
            const float* Aa = Aqkv + (size_t)(l * 3 + kk) * R_ * H_;
            const float* Bb = Bqkv + (size_t)(l * 3 + kk) * H_ * R_;
            gemm(stream, hnb, W,  nullptr, qks[kk], nullptr,
                 H_, H_, H_, H_, H_, H_, 1, 1.f, GF_WRITEF);
            gemm(stream, hnb, Aa, nullptr, nullptr, tmpb,
                 R_, H_, H_, H_, RP_, 1, H_, 1.f, GF_OUTB);
            gemm(stream, tmpb, Bb, nullptr, qks[kk], nullptr,
                 H_, RP_, R_, RP_, H_, 1, R_, LORA_SCALE_, GF_ACC | GF_WRITEF);
        }
        k_rope<<<M_, 256, 0, stream>>>(qk0, qk1, qk2, qb, kbuf, vbuf);
        k_attn<<<dim3(T_ / 128, NH_, B_), 256, 0, stream>>>(qb, kbuf, vbuf, attb);

        gemm(stream, attb, Wo + (size_t)l * H_ * H_, nullptr, h, nullptr,
             H_, H_, H_, H_, H_, H_, 1, 1.f, GF_ACC | GF_WRITEF);
        gemm(stream, attb, Ao + (size_t)l * R_ * H_, nullptr, nullptr, tmpb,
             R_, H_, H_, H_, RP_, 1, H_, 1.f, GF_OUTB);
        gemm(stream, tmpb, Bo + (size_t)l * H_ * R_, nullptr, h, nullptr,
             H_, RP_, R_, RP_, H_, 1, R_, LORA_SCALE_, GF_ACC | GF_WRITEF);

        k_rms<<<M_, 256, 0, stream>>>(h, ln2 + (size_t)l * H_, hnb);
        for (int kk = 0; kk < 2; ++kk) {
            const float* W  = Wgu + (size_t)(l * 2 + kk) * H_ * F_;
            const float* Aa = Agu + (size_t)(l * 2 + kk) * R_ * H_;
            const float* Bb = Bgu + (size_t)(l * 2 + kk) * F_ * R_;
            gemm(stream, hnb, W,  nullptr, gus[kk], nullptr,
                 F_, H_, H_, H_, F_, F_, 1, 1.f, GF_WRITEF);
            gemm(stream, hnb, Aa, nullptr, nullptr, tmpb,
                 R_, H_, H_, H_, RP_, 1, H_, 1.f, GF_OUTB);
            gemm(stream, tmpb, Bb, nullptr, gus[kk], nullptr,
                 F_, RP_, R_, RP_, F_, 1, R_, LORA_SCALE_, GF_ACC | GF_WRITEF);
        }
        {
            const int n = M_ * F_;
            k_silumul<<<(n + 255) / 256, 256, 0, stream>>>(gu0, gu1, actb, n);
        }
        gemm(stream, actb, Wd + (size_t)l * F_ * H_, nullptr, h, nullptr,
             H_, F_, F_, F_, H_, H_, 1, 1.f, GF_ACC | GF_WRITEF);
        gemm(stream, actb, Ad + (size_t)l * R_ * F_, nullptr, nullptr, tmpb,
             R_, F_, F_, F_, RP_, 1, F_, 1.f, GF_OUTB);
        gemm(stream, tmpb, Bd + (size_t)l * H_ * R_, nullptr, h, nullptr,
             H_, RP_, R_, RP_, H_, 1, R_, LORA_SCALE_, GF_ACC | GF_WRITEF);
    }

    k_rms<<<M_, 256, 0, stream>>>(h, lnf, hnb);
    gemm(stream, hnb, headW, headb, logits, nullptr,
         NC_, H_, H_, H_, NC_, NC_, 1, 1.f, GF_WRITEF | GF_BIAS);
    k_head<<<M_, 32, 0, stream>>>(logits, temp, (float*)d_out);
}
